// Attention_18631568130094
// MI455X (gfx1250) — matrix operand order repaired and checked by reading, not compiled, not run on hardware
//
#include <hip/hip_runtime.h>
#include <hip/hip_bf16.h>

// Cosine-similarity attention scores, CDNA5 (gfx1250, wave32).
//   scores[b,q,k] = <Qn[b,q,:], Kn[b,k,:]>  (rows pre-normalized), masked to -1e9.
// Kernel 1 normalizes Q/K rows into f16 workspace (32 MB of d_ws).
// Kernel 2: 128x128-tile GEMM, 4 waves, 64x64 wave tile (16 WMMAs/K-step),
// global->LDS staging via GLOBAL_LOAD_ASYNC_TO_LDS_B128 (ASYNCcnt),
// conflict-free LDS stride (112 B = 28 banks). K-loop kept at unroll(1) and
// A-fragments serialized to stay under 256 VGPRs (no spills, no MSB mode).

typedef __attribute__((ext_vector_type(16))) _Float16 v16h;
typedef __attribute__((ext_vector_type(4)))  _Float16 v4h;
typedef __attribute__((ext_vector_type(8)))  float    v8f;

static constexpr int   B_  = 4;
static constexpr int   SQ  = 2048;
static constexpr int   SK  = 2048;
static constexpr int   D_  = 1024;
static constexpr float NEGV = -1000000000.0f;

static constexpr int BM  = 128;   // block tile M
static constexpr int BN  = 128;   // block tile N
static constexpr int BK  = 32;    // K chunk per WMMA step (f16 K=32)
static constexpr int LDK = 56;    // LDS row stride in halves: 112 B = 28 banks,
                                  // 16B-aligned and conflict-free across 16 rows
static constexpr int NK  = D_ / BK;

union Frag {
    v16h h;
    int4 q[2];
};

// 64-byte async DMA global -> LDS (one tile row of 32 halves).
// INST_OFFSET is added to BOTH the LDS and global addresses (ISA 08 §4.4),
// so a single address pair serves all four 16 B beats.
__device__ __forceinline__ void async_copy64(unsigned ldsAddr, const void* gptr)
{
    unsigned long long ga = (unsigned long long)gptr;
    asm volatile("global_load_async_to_lds_b128 %0, %1, off\n\t"
                 "global_load_async_to_lds_b128 %0, %1, off offset:16\n\t"
                 "global_load_async_to_lds_b128 %0, %1, off offset:32\n\t"
                 "global_load_async_to_lds_b128 %0, %1, off offset:48"
                 :: "v"(ldsAddr), "v"(ga) : "memory");
}

__device__ __forceinline__ void wait_async0()
{
    asm volatile("s_wait_asynccnt 0x0" ::: "memory");
}

// ---------------------------------------------------------------------------
// Kernel 1: row L2-normalize + convert to f16.
// ---------------------------------------------------------------------------
__global__ __launch_bounds__(256)
void prep_norm_f16(const float* __restrict__ q,
                   const float* __restrict__ k,
                   _Float16* __restrict__ qh,
                   _Float16* __restrict__ kh,
                   int rowsPerMat)
{
    const int row = blockIdx.x;
    const float* src;
    _Float16*    dst;
    if (row < rowsPerMat) {
        src = q  + (size_t)row * D_;
        dst = qh + (size_t)row * D_;
    } else {
        const int r2 = row - rowsPerMat;
        src = k  + (size_t)r2 * D_;
        dst = kh + (size_t)r2 * D_;
    }

    const int t = threadIdx.x;                    // 256 threads, 4 floats each
    const float4 v = ((const float4*)src)[t];
    float ss = v.x * v.x + v.y * v.y + v.z * v.z + v.w * v.w;

    __shared__ float red[256];
    red[t] = ss;
    __syncthreads();
#pragma unroll
    for (int s = 128; s > 0; s >>= 1) {
        if (t < s) red[t] += red[t + s];
        __syncthreads();
    }
    const float inv = 1.0f / fmaxf(sqrtf(red[0]), 1e-8f);

    v4h h;
    h.x = (_Float16)(v.x * inv);
    h.y = (_Float16)(v.y * inv);
    h.z = (_Float16)(v.z * inv);
    h.w = (_Float16)(v.w * inv);
    ((v4h*)dst)[t] = h;
}

// ---------------------------------------------------------------------------
// Kernel 2: per-batch GEMM  out[b] = Qn[b] (2048x1024) * Kn[b]^T (1024x2048)
// Block: 128x128 tile, 128 threads (4 waves as 2(M) x 2(N)).
// Wave: 64x64 sub-tile = 4x4 WMMA accumulators (v8f each).
// ---------------------------------------------------------------------------
__global__ __launch_bounds__(128)
void cosine_scores_wmma(const _Float16* __restrict__ qh,
                        const _Float16* __restrict__ kh,
                        const int* __restrict__ mask,
                        float* __restrict__ out)
{
    __shared__ __align__(16) _Float16 ldsA[2][BM * LDK];   // 2 x 14 KB
    __shared__ __align__(16) _Float16 ldsB[2][BN * LDK];   // 2 x 14 KB

    const int b     = blockIdx.z;
    const int mBase = blockIdx.x * BM;
    const int nBase = blockIdx.y * BN;
    const int tid   = threadIdx.x;
    const int lane  = tid & 31;
    const int wave  = tid >> 5;
    const int waveM = wave & 1;    // 0..1 -> 64-row halves
    const int waveN = wave >> 1;   // 0..1 -> 64-col halves

    // Each thread DMAs one 32-half tile row (64 B) of A and of B per K chunk.
    const _Float16* aGlob = qh + ((size_t)b * SQ + mBase + tid) * D_;
    const _Float16* bGlob = kh + ((size_t)b * SK + nBase + tid) * D_;

    const unsigned aLds0 = (unsigned)(uintptr_t)&ldsA[0][tid * LDK];
    const unsigned aLds1 = (unsigned)(uintptr_t)&ldsA[1][tid * LDK];
    const unsigned bLds0 = (unsigned)(uintptr_t)&ldsB[0][tid * LDK];
    const unsigned bLds1 = (unsigned)(uintptr_t)&ldsB[1][tid * LDK];

    // prologue: DMA chunk 0 into buffer 0
    async_copy64(aLds0, aGlob);
    async_copy64(bLds0, bGlob);
    wait_async0();
    __syncthreads();

    const v8f vzero = {};
    v8f acc[4][4];
#pragma unroll
    for (int mt = 0; mt < 4; ++mt)
#pragma unroll
        for (int nt = 0; nt < 4; ++nt)
            acc[mt][nt] = vzero;

    // Fragment addressing (ISA 7.12.2, 16-bit layouts):
    //  A 16x32: lane<16 holds row=lane, halves {K0..7, K16..23};
    //           lane>=16 holds row=lane-16, halves {K8..15, K24..31}.
    //  B 32x16: lane<16 holds col=lane, halves {K0..7, K16..23};
    //           lane>=16 holds col=lane-16, halves {K8..15, K24..31}.
    const int fr    = lane & 15;
    const int aKoff = (lane >> 4) * 8;    // 0 or 8 halves
    const int bKoff = (lane >> 4) * 8;    // 0 or 8 halves

    const int aRow0 = (waveM * 64 + fr) * LDK;   // + mt*16*LDK per m-tile
    const int bCol0 = (waveN * 64 + fr) * LDK;   // + nt*16*LDK per n-tile

#pragma unroll 1   // keep ONE fragment set live: no unroll-by-2 register blowup
    for (int kb = 0; kb < NK; ++kb) {
        const int buf = kb & 1;

        // DMA next chunk into the other buffer; in flight during the WMMAs.
        if (kb + 1 < NK) {
            async_copy64(buf ? aLds0 : aLds1, aGlob + (kb + 1) * BK);
            async_copy64(buf ? bLds0 : bLds1, bGlob + (kb + 1) * BK);
        }

        const _Float16* aB = ldsA[buf];
        const _Float16* bB = ldsB[buf];

        // All 4 B fragments stay live (32 VGPRs)...
        Frag bf[4];
#pragma unroll
        for (int nt = 0; nt < 4; ++nt) {
            const int c = bCol0 + nt * 16 * LDK;
            bf[nt].q[0] = *(const int4*)&bB[c + bKoff];
            bf[nt].q[1] = *(const int4*)&bB[c + 16 + bKoff];
        }
        // ...but only one A fragment at a time (8 VGPRs).
#pragma unroll
        for (int mt = 0; mt < 4; ++mt) {
            Frag af;
            const int r = aRow0 + mt * 16 * LDK;
            af.q[0] = *(const int4*)&aB[r + aKoff];
            af.q[1] = *(const int4*)&aB[r + 16 + aKoff];
#pragma unroll
            for (int nt = 0; nt < 4; ++nt)
                acc[mt][nt] = __builtin_amdgcn_wmma_f32_16x16x32_f16(
                    false, af.h, false, bf[nt].h,
                    (short)0, acc[mt][nt], false, false);
        }

        if (kb + 1 < NK) wait_async0();   // own DMAs into buf^1 complete
        __syncthreads();                  // visible to all waves
    }

    // Epilogue: C layout (ISA 7.12.2): VGPR v -> row v (lanes 0-15) / row v+8
    // (lanes 16-31); col = lane&15. Apply column mask, write fp32.
    const int rowSel = (lane >> 4) * 8;
#pragma unroll
    for (int nt = 0; nt < 4; ++nt) {
        const int col    = nBase + waveN * 64 + nt * 16 + fr;
        const bool valid = mask[b * SK + col] != 0;
#pragma unroll
        for (int mt = 0; mt < 4; ++mt) {
            const int rbase = mBase + waveM * 64 + mt * 16 + rowSel;
            float* op = out + ((size_t)b * SQ + rbase) * SK + col;
            const v8f a = acc[mt][nt];
#pragma unroll
            for (int v = 0; v < 8; ++v)
                op[(size_t)v * SK] = valid ? a[v] : NEGV;
        }
    }
}

// ---------------------------------------------------------------------------
extern "C" void kernel_launch(void* const* d_in, const int* in_sizes, int n_in,
                              void* d_out, int out_size, void* d_ws, size_t ws_size,
                              hipStream_t stream)
{
    const float* q    = (const float*)d_in[0];   // (B, Sq, 1, D) fp32
    const float* k    = (const float*)d_in[1];   // (B, 1, Sk, D) fp32
    const int*   mask = (const int*)d_in[2];     // (B, Sk) int32
    float*       out  = (float*)d_out;           // (B, Sq, Sk) fp32

    _Float16* qh = (_Float16*)d_ws;                       // 16 MB
    _Float16* kh = qh + (size_t)B_ * SQ * D_;             // +16 MB (needs 32 MB ws)

    const int rowsPerMat = B_ * SQ;
    prep_norm_f16<<<2 * rowsPerMat, 256, 0, stream>>>(q, k, qh, kh, rowsPerMat);

    dim3 grid(SQ / BM, SK / BN, B_);
    cosine_scores_wmma<<<grid, 128, 0, stream>>>(qh, kh, mask, out);
}
